// TransformerEncoderLayer_420906795677
// MI455X (gfx1250) — matrix operand order repaired and checked by reading, not compiled, not run on hardware
//
#include <hip/hip_runtime.h>

// ---------------------------------------------------------------------------
// Types for CDNA5 WMMA / TDM
// ---------------------------------------------------------------------------
typedef __bf16 bf16_t;
typedef __attribute__((ext_vector_type(16))) __bf16 v16bf;
typedef __attribute__((ext_vector_type(8)))  float  v8f;
typedef __attribute__((ext_vector_type(4)))  unsigned int u32x4;
typedef __attribute__((ext_vector_type(8)))  int i32x8;
typedef __attribute__((ext_vector_type(4)))  int i32x4;

union Frag16 {              // one WMMA 16-bit operand fragment: 8 VGPRs
    v16bf v;
    uint4 q[2];
};

__device__ __forceinline__ v8f wmma_bf16(const Frag16& a, const Frag16& b, v8f c) {
    return __builtin_amdgcn_wmma_f32_16x16x32_bf16(
        false, a.v, false, b.v, (short)0, c, false, false);
}

#if __has_builtin(__builtin_amdgcn_tensor_load_to_lds)
#define USE_TDM 1
#else
#define USE_TDM 0
#endif

// Low 32 bits of a flat LDS pointer = LDS byte offset (ISA 10.2 aperture rules)
template <typename T>
__device__ __forceinline__ unsigned lds_addr_of(T* p) {
    return (unsigned)(unsigned long long)(void*)p;
}

#if USE_TDM
// Issue a 2D TDM load: tile (tile_cols x tile_rows) of 2-byte elements from a
// row-major tensor with row stride kd_elems into LDS at lds_addr.
// D# packing per CDNA5 ISA 8.3 (group 0) / 8.4 (group 1).
__device__ __forceinline__ void tdm_load_2d(unsigned lds_addr, const void* gaddr,
                                            int rows_total, int kd_elems,
                                            int tile_rows, int tile_cols) {
    unsigned long long ga = (unsigned long long)gaddr;
    u32x4 g0;
    g0[0] = 1u;                                            // count=1 (valid), user mode
    g0[1] = lds_addr;                                      // lds_addr [63:32]
    g0[2] = (unsigned)(ga & 0xFFFFFFFFu);                  // global_addr [95:64]
    g0[3] = (unsigned)((ga >> 32) & 0x01FFFFFFu)           // global_addr [120:96]
          | (2u << 30);                                    // type=2 ("image") [127:126]
    i32x8 g1;
    g1[0] = (int)(1u << 16);                               // data_size=1 (2 bytes)
    g1[1] = (int)(((unsigned)kd_elems & 0xFFFFu) << 16);   // tensor_dim0 lo @bit48
    g1[2] = (int)((((unsigned)kd_elems >> 16) & 0xFFFFu)   // tensor_dim0 hi
          | (((unsigned)rows_total & 0xFFFFu) << 16));     // tensor_dim1 lo @bit80
    g1[3] = (int)((((unsigned)rows_total >> 16) & 0xFFFFu) // tensor_dim1 hi
          | ((unsigned)tile_cols << 16));                  // tile_dim0 @bit112
    g1[4] = (int)((unsigned)tile_rows);                    // tile_dim1 (tile_dim2=0)
    g1[5] = (int)((unsigned)kd_elems);                     // tensor_dim0_stride lo32
    g1[6] = 0;                                             // stride0 hi / dim1_stride lo
    g1[7] = 0;                                             // dim1_stride hi
    i32x4 z4 = {0, 0, 0, 0};
#if defined(__clang_major__) && (__clang_major__ >= 23)
    i32x8 z8 = {0, 0, 0, 0, 0, 0, 0, 0};
    __builtin_amdgcn_tensor_load_to_lds(g0, g1, z4, z4, z8, 0);
#else
    __builtin_amdgcn_tensor_load_to_lds(g0, g1, z4, z4, 0);
#endif
}
#endif

// Problem constants
#define Bn 2
#define Sn 2048
#define Dn 1024
#define Hn 16
#define HDn 64
#define FFn 4096
#define Mtok (Bn * Sn)   // 4096 token rows

// ---------------------------------------------------------------------------
// Weight convert+transpose:  W fp32 [K][N]  ->  Wt bf16 [N][K]
// ---------------------------------------------------------------------------
__global__ __launch_bounds__(256)
void wt_convert(const float* __restrict__ W, bf16_t* __restrict__ Wt, int K, int N) {
    __shared__ float tile[32][33];
    int n0 = blockIdx.x * 32, k0 = blockIdx.y * 32;
    int tx = threadIdx.x, ty = threadIdx.y;          // block (32, 8)
#pragma unroll
    for (int i = 0; i < 32; i += 8)
        tile[ty + i][tx] = W[(size_t)(k0 + ty + i) * N + n0 + tx];
    __syncthreads();
#pragma unroll
    for (int i = 0; i < 32; i += 8)
        Wt[(size_t)(n0 + ty + i) * K + k0 + tx] = (bf16_t)tile[tx][ty + i];
}

// ---------------------------------------------------------------------------
// LayerNorm: fp32 row [D] -> bf16 row [D]
// ---------------------------------------------------------------------------
__global__ __launch_bounds__(256)
void ln_bf16(const float* __restrict__ x, const float* __restrict__ g,
             const float* __restrict__ b, bf16_t* __restrict__ out) {
    const int row = blockIdx.x;
    const int tid = threadIdx.x;
    const float* xr = x + (size_t)row * Dn;
    float v[4], s = 0.f, s2 = 0.f;
#pragma unroll
    for (int i = 0; i < 4; ++i) {
        v[i] = xr[tid + i * 256];
        s  += v[i];
        s2 += v[i] * v[i];
    }
    __shared__ float r1[256], r2[256];
    r1[tid] = s; r2[tid] = s2;
    __syncthreads();
    for (int off = 128; off > 0; off >>= 1) {
        if (tid < off) { r1[tid] += r1[tid + off]; r2[tid] += r2[tid + off]; }
        __syncthreads();
    }
    float mu  = r1[0] * (1.0f / Dn);
    float var = r2[0] * (1.0f / Dn) - mu * mu;
    float rs  = rsqrtf(var + 1e-5f);
#pragma unroll
    for (int i = 0; i < 4; ++i) {
        int c = tid + i * 256;
        out[(size_t)row * Dn + c] = (bf16_t)((v[i] - mu) * rs * g[c] + b[c]);
    }
}

// ---------------------------------------------------------------------------
// Tiled WMMA GEMM with double-buffered TDM (tensor_load_to_lds) staging.
// C[M,N] = A[M,K](bf16,row) * Bt[N,K](bf16,row-per-N) + bias.
// Block tile 128x128, 8 waves (2 x 4), wave tile 64x32 (4x2 WMMA accums).
// ---------------------------------------------------------------------------
template <bool GELU, bool RES, bool OUT32, bool OUT16>
__global__ __launch_bounds__(256)
void gemm_bf16(const bf16_t* __restrict__ A, const bf16_t* __restrict__ Bt,
               const float* __restrict__ bias, const float* __restrict__ res,
               float* __restrict__ out32, bf16_t* __restrict__ out16,
               int M, int N, int Kd) {
    __shared__ bf16_t As[2][128][32];
    __shared__ bf16_t Bs[2][128][32];

    const int tid  = threadIdx.x;
    const int lane = tid & 31;
    const int wave = tid >> 5;
    const int wm   = wave >> 2;          // 0..1 : 64 rows each
    const int wn   = wave & 3;           // 0..3 : 32 cols each
    const int n16  = lane & 15;
    const int g    = lane >> 4;
    const int m0   = blockIdx.y * 128;
    const int n0   = blockIdx.x * 128;
    const int nk   = Kd >> 5;

    v8f acc[4][2];
#pragma unroll
    for (int am = 0; am < 4; ++am)
#pragma unroll
        for (int bn = 0; bn < 2; ++bn) acc[am][bn] = (v8f){};

#if USE_TDM
    if (wave == 0) {   // prologue: async-stage slab 0 into buffer 0
        tdm_load_2d(lds_addr_of(&As[0][0][0]), A  + (size_t)m0 * Kd, M, Kd, 128, 32);
        tdm_load_2d(lds_addr_of(&Bs[0][0][0]), Bt + (size_t)n0 * Kd, N, Kd, 128, 32);
    }
#endif

    for (int kt = 0; kt < nk; ++kt) {
        const int cur = kt & 1;
#if USE_TDM
        __builtin_amdgcn_s_wait_tensorcnt(0);   // wave0: slab kt landed (others: no-op)
        __syncthreads();                        // everyone may read buffer `cur`
        if (wave == 0 && kt + 1 < nk) {         // prefetch slab kt+1 into other buffer
            int k0n = (kt + 1) << 5;
            tdm_load_2d(lds_addr_of(&As[cur ^ 1][0][0]),
                        A  + (size_t)m0 * Kd + k0n, M, Kd, 128, 32);
            tdm_load_2d(lds_addr_of(&Bs[cur ^ 1][0][0]),
                        Bt + (size_t)n0 * Kd + k0n, N, Kd, 128, 32);
        }
#else
        {   // fallback: cooperative vector-load staging
            int k0 = kt << 5;
#pragma unroll
            for (int i = 0; i < 2; ++i) {
                int idx = tid * 8 + i * 2048;
                int r = idx >> 5, c = idx & 31;
                *(uint4*)&As[cur][r][c] = *(const uint4*)(A  + (size_t)(m0 + r) * Kd + k0 + c);
                *(uint4*)&Bs[cur][r][c] = *(const uint4*)(Bt + (size_t)(n0 + r) * Kd + k0 + c);
            }
            __syncthreads();
        }
#endif
        Frag16 af[4], bf[2];
#pragma unroll
        for (int am = 0; am < 4; ++am) {
            int row = wm * 64 + am * 16 + n16;
            af[am].q[0] = *(const uint4*)&As[cur][row][8 * g];       // K = 8g..8g+7
            af[am].q[1] = *(const uint4*)&As[cur][row][16 + 8 * g];  // K = 16+8g..
        }
#pragma unroll
        for (int bn = 0; bn < 2; ++bn) {
            int row = wn * 32 + bn * 16 + n16;
            bf[bn].q[0] = *(const uint4*)&Bs[cur][row][8 * g];       // K = 8g..8g+7
            bf[bn].q[1] = *(const uint4*)&Bs[cur][row][16 + 8 * g];  // K = 16+8g..
        }
#pragma unroll
        for (int am = 0; am < 4; ++am)
#pragma unroll
            for (int bn = 0; bn < 2; ++bn)
                acc[am][bn] = wmma_bf16(af[am], bf[bn], acc[am][bn]);
#if !USE_TDM
        __syncthreads();
#endif
    }

#pragma unroll
    for (int am = 0; am < 4; ++am)
#pragma unroll
        for (int bn = 0; bn < 2; ++bn) {
            int col = n0 + wn * 32 + bn * 16 + n16;
            float bv = bias[col];
#pragma unroll
            for (int r = 0; r < 8; ++r) {
                int row = m0 + wm * 64 + am * 16 + r + 8 * g;
                float v = acc[am][bn][r] + bv;
                if (GELU) v = 0.5f * v * (1.0f + erff(v * 0.70710678118f));
                if (RES)  v += res[(size_t)row * N + col];
                if (OUT32) out32[(size_t)row * N + col] = v;
                if (OUT16) out16[(size_t)row * N + col] = (bf16_t)v;
            }
        }
}

// ---------------------------------------------------------------------------
// Flash attention (non-causal). 8 waves/block share one (b,h); each wave owns
// 16 query rows; key tiles of 32. The V^T tile is staged once per block.
// grid = (S/128, B*H), block = 256.
// ---------------------------------------------------------------------------
__global__ __launch_bounds__(256)
void attn_flash(const bf16_t* __restrict__ Q, const bf16_t* __restrict__ Kmat,
                const bf16_t* __restrict__ V, bf16_t* __restrict__ ctx) {
    __shared__ bf16_t P_lds[8][16][32];    // 8 KB: per-wave P tile
    __shared__ bf16_t Vt_lds[64][32];      // 4 KB: shared V^T tile [d][key]

    const int tid  = threadIdx.x;
    const int lane = tid & 31;
    const int wave = tid >> 5;
    const int n16  = lane & 15;
    const int g    = lane >> 4;
    const int b    = blockIdx.y >> 4;
    const int h    = blockIdx.y & 15;
    const int q0   = b * Sn + blockIdx.x * 128 + wave * 16;
    const int c0   = h * HDn;
    const float scale = 0.125f;            // 1/sqrt(64)

    // Q fragments (2 K-slabs of 32 over HD=64), loaded once
    Frag16 qf[2];
    {
        const bf16_t* qp = Q + (size_t)(q0 + n16) * Dn + c0;
#pragma unroll
        for (int kk = 0; kk < 2; ++kk) {
            qf[kk].q[0] = *(const uint4*)(qp + kk * 32 + 8 * g);
            qf[kk].q[1] = *(const uint4*)(qp + kk * 32 + 16 + 8 * g);
        }
    }

    v8f O[4];
#pragma unroll
    for (int dt = 0; dt < 4; ++dt) O[dt] = (v8f){};
    float mrow[8], lrow[8];
#pragma unroll
    for (int r = 0; r < 8; ++r) { mrow[r] = -1e30f; lrow[r] = 0.f; }

    for (int kt = 0; kt < Sn / 32; ++kt) {
        const int krow = b * Sn + kt * 32;

        __syncthreads();   // previous tile's PV reads of Vt_lds are done

        // Cooperative V^T staging: thread -> (key = tid&31, 8 d values)
        {
            int key = tid & 31;
            int dc  = (tid >> 5) * 8;
            union { uint4 u; bf16_t e[8]; } cv;
            cv.u = *(const uint4*)(V + (size_t)(krow + key) * Dn + c0 + dc);
#pragma unroll
            for (int j = 0; j < 8; ++j) Vt_lds[dc + j][key] = cv.e[j];
        }

        // K fragments: B-frag layout == rows of K ([key][d]) loaded directly
        Frag16 kf[2][2];
#pragma unroll
        for (int half = 0; half < 2; ++half) {
            const bf16_t* kp = Kmat + (size_t)(krow + half * 16 + n16) * Dn + c0;
#pragma unroll
            for (int kk = 0; kk < 2; ++kk) {
                kf[half][kk].q[0] = *(const uint4*)(kp + kk * 32 + 8 * g);
                kf[half][kk].q[1] = *(const uint4*)(kp + kk * 32 + 16 + 8 * g);
            }
        }

        v8f s0 = (v8f){}, s1 = (v8f){};
        s0 = wmma_bf16(qf[0], kf[0][0], s0);
        s0 = wmma_bf16(qf[1], kf[0][1], s0);   // keys  0..15 of tile
        s1 = wmma_bf16(qf[0], kf[1][0], s1);
        s1 = wmma_bf16(qf[1], kf[1][1], s1);   // keys 16..31 of tile

        // Online softmax. C-layout: accum elem r of lane-group g is row r+8g.
        float sc0[8], sc1[8], alpha[8];
#pragma unroll
        for (int r = 0; r < 8; ++r) { sc0[r] = s0[r] * scale; sc1[r] = s1[r] * scale; }
#pragma unroll
        for (int r = 0; r < 8; ++r) {
            float t = fmaxf(sc0[r], sc1[r]);
#pragma unroll
            for (int off = 1; off < 16; off <<= 1)
                t = fmaxf(t, __shfl_xor(t, off, 16));   // row-max over 16 lanes
            float nm = fmaxf(mrow[r], t);
            alpha[r] = __expf(mrow[r] - nm);
            mrow[r]  = nm;
        }
#pragma unroll
        for (int dt = 0; dt < 4; ++dt)
#pragma unroll
            for (int r = 0; r < 8; ++r) O[dt][r] *= alpha[r];

#pragma unroll
        for (int r = 0; r < 8; ++r) {
            float p0 = __expf(sc0[r] - mrow[r]);
            float p1 = __expf(sc1[r] - mrow[r]);
            float t  = p0 + p1;
#pragma unroll
            for (int off = 1; off < 16; off <<= 1)
                t += __shfl_xor(t, off, 16);            // row-sum
            lrow[r] = lrow[r] * alpha[r] + t;
            P_lds[wave][r + 8 * g][n16]      = (bf16_t)p0;
            P_lds[wave][r + 8 * g][16 + n16] = (bf16_t)p1;
        }
        __builtin_amdgcn_sched_barrier(0);  // keep per-wave DS stores before DS loads

        // Reload P as an A-fragment (16x32 over keys); same-wave DS is in-order.
        Frag16 pf;
        pf.q[0] = *(const uint4*)&P_lds[wave][n16][8 * g];
        pf.q[1] = *(const uint4*)&P_lds[wave][n16][16 + 8 * g];

        __syncthreads();   // shared V^T tile fully staged

#pragma unroll
        for (int dt = 0; dt < 4; ++dt) {
            Frag16 vf;
            vf.q[0] = *(const uint4*)&Vt_lds[dt * 16 + n16][8 * g];
            vf.q[1] = *(const uint4*)&Vt_lds[dt * 16 + n16][16 + 8 * g];
            O[dt] = wmma_bf16(pf, vf, O[dt]);
        }
    }

    // Finalize: O / l, write ctx (bf16, token-major, head columns)
#pragma unroll
    for (int dt = 0; dt < 4; ++dt)
#pragma unroll
        for (int r = 0; r < 8; ++r) {
            float val = O[dt][r] / lrow[r];
            ctx[(size_t)(q0 + r + 8 * g) * Dn + c0 + dt * 16 + n16] = (bf16_t)val;
        }
}

// ---------------------------------------------------------------------------
// Launch
// ---------------------------------------------------------------------------
extern "C" void kernel_launch(void* const* d_in, const int* in_sizes, int n_in,
                              void* d_out, int out_size, void* d_ws, size_t ws_size,
                              hipStream_t stream) {
    (void)in_sizes; (void)n_in; (void)out_size; (void)ws_size;

    const float* x    = (const float*)d_in[0];
    const float* Wq   = (const float*)d_in[1];
    const float* bq   = (const float*)d_in[2];
    const float* Wk   = (const float*)d_in[3];
    const float* bk   = (const float*)d_in[4];
    const float* Wv   = (const float*)d_in[5];
    const float* bv   = (const float*)d_in[6];
    const float* Wo   = (const float*)d_in[7];
    const float* bo   = (const float*)d_in[8];
    const float* ln1g = (const float*)d_in[9];
    const float* ln1b = (const float*)d_in[10];
    const float* W1   = (const float*)d_in[11];
    const float* b1   = (const float*)d_in[12];
    const float* W2   = (const float*)d_in[13];
    const float* b2   = (const float*)d_in[14];
    const float* ln2g = (const float*)d_in[15];
    const float* ln2b = (const float*)d_in[16];
    float* out = (float*)d_out;

    char* ws = (char*)d_ws;
    size_t off = 0;
    auto carve = [&](size_t bytes) -> void* {
        void* p = ws + off;
        off += (bytes + 255) & ~(size_t)255;
        return p;
    };
    bf16_t* WqT = (bf16_t*)carve((size_t)Dn * Dn * 2);
    bf16_t* WkT = (bf16_t*)carve((size_t)Dn * Dn * 2);
    bf16_t* WvT = (bf16_t*)carve((size_t)Dn * Dn * 2);
    bf16_t* WoT = (bf16_t*)carve((size_t)Dn * Dn * 2);
    bf16_t* W1T = (bf16_t*)carve((size_t)Dn * FFn * 2);  // [FF][D]
    bf16_t* W2T = (bf16_t*)carve((size_t)FFn * Dn * 2);  // [D][FF]
    bf16_t* xn  = (bf16_t*)carve((size_t)Mtok * Dn * 2);
    bf16_t* Qb  = (bf16_t*)carve((size_t)Mtok * Dn * 2);
    bf16_t* Kb  = (bf16_t*)carve((size_t)Mtok * Dn * 2);
    bf16_t* Vb  = (bf16_t*)carve((size_t)Mtok * Dn * 2);
    bf16_t* ctx = (bf16_t*)carve((size_t)Mtok * Dn * 2);
    float*  x1  = (float*) carve((size_t)Mtok * Dn * 4);
    bf16_t* hb  = (bf16_t*)carve((size_t)Mtok * Dn * 2);
    bf16_t* act = (bf16_t*)carve((size_t)Mtok * FFn * 2);

    dim3 tb(32, 8);
    wt_convert<<<dim3(Dn / 32, Dn / 32),  tb, 0, stream>>>(Wq, WqT, Dn, Dn);
    wt_convert<<<dim3(Dn / 32, Dn / 32),  tb, 0, stream>>>(Wk, WkT, Dn, Dn);
    wt_convert<<<dim3(Dn / 32, Dn / 32),  tb, 0, stream>>>(Wv, WvT, Dn, Dn);
    wt_convert<<<dim3(Dn / 32, Dn / 32),  tb, 0, stream>>>(Wo, WoT, Dn, Dn);
    wt_convert<<<dim3(FFn / 32, Dn / 32), tb, 0, stream>>>(W1, W1T, Dn, FFn);
    wt_convert<<<dim3(Dn / 32, FFn / 32), tb, 0, stream>>>(W2, W2T, FFn, Dn);

    ln_bf16<<<Mtok, 256, 0, stream>>>(x, ln1g, ln1b, xn);

    dim3 gD(Dn / 128, Mtok / 128);
    gemm_bf16<false, false, false, true><<<gD, 256, 0, stream>>>(
        xn, WqT, bq, nullptr, nullptr, Qb, Mtok, Dn, Dn);
    gemm_bf16<false, false, false, true><<<gD, 256, 0, stream>>>(
        xn, WkT, bk, nullptr, nullptr, Kb, Mtok, Dn, Dn);
    gemm_bf16<false, false, false, true><<<gD, 256, 0, stream>>>(
        xn, WvT, bv, nullptr, nullptr, Vb, Mtok, Dn, Dn);

    attn_flash<<<dim3(Sn / 128, Bn * Hn), 256, 0, stream>>>(Qb, Kb, Vb, ctx);

    gemm_bf16<false, true, true, false><<<gD, 256, 0, stream>>>(
        ctx, WoT, bo, x, x1, nullptr, Mtok, Dn, Dn);

    ln_bf16<<<Mtok, 256, 0, stream>>>(x1, ln2g, ln2b, hb);

    gemm_bf16<true, false, false, true><<<dim3(FFn / 128, Mtok / 128), 256, 0, stream>>>(
        hb, W1T, b1, nullptr, nullptr, act, Mtok, FFn, Dn);

    gemm_bf16<false, true, true, false><<<gD, 256, 0, stream>>>(
        act, W2T, b2, x1, out, nullptr, Mtok, Dn, FFn);
}
